// Lorentz_grapher_42580305773091
// MI455X (gfx1250) — matrix operand order repaired and checked by reading, not compiled, not run on hardware
//
#include <hip/hip_runtime.h>
#include <math.h>

// Problem constants (from setup_inputs): x = (8, 192, 56, 56)
#define BATCH 8
#define CH    192
#define HW    3136
#define NPT   3136        // H*W tokens per batch
#define KNN   9

typedef __attribute__((ext_vector_type(2)))  float  v2f;
typedef __attribute__((ext_vector_type(8)))  float  v8f;
typedef __attribute__((ext_vector_type(16))) __bf16 v16bf;

__device__ __forceinline__ float gelu_tanh(float x) {
  // jax.nn.gelu default (approximate=True, tanh form)
  float inner = 0.7978845608028654f * (x + 0.044715f * x * x * x);
  return 0.5f * x * (1.0f + tanhf(inner));
}

// ---------------------------------------------------------------------------
// 1) (B, C, HW) -> (B, HW, C) tiled transpose
// ---------------------------------------------------------------------------
__global__ void __launch_bounds__(256)
transpose_in_kernel(const float* __restrict__ x, float* __restrict__ xn) {
  __shared__ float tile[16][17];
  int hw0 = blockIdx.x * 16, c0 = blockIdx.y * 16, b = blockIdx.z;
  int tx = threadIdx.x, ty = threadIdx.y;
  tile[ty][tx] = x[((size_t)b * CH + c0 + ty) * HW + hw0 + tx];
  __syncthreads();
  xn[((size_t)b * HW + hw0 + ty) * CH + c0 + tx] = tile[tx][ty];
}

// ---------------------------------------------------------------------------
// 2) Fused lorentz_linear: out = lorentz(gelu(Xin) @ W^T + bias) [+ res]
//    Xin: (R, Cin), W: (192, Cin), out: (R, 192).  One block = 16 rows,
//    128 threads = 4 waves, each wave does 3 of the 12 16-col tiles with
//    V_WMMA_F32_16X16X4_F32 (exact fp32), epilogue fused via LDS.
// ---------------------------------------------------------------------------
#define ASTR 386   // Cin(max 384)+2 pad: conflict-free float2 LDS reads
#define YSTR 193

__global__ void __launch_bounds__(128)
lorentz_linear_kernel(const float* __restrict__ Xin,
                      const float* __restrict__ Wt,
                      const float* __restrict__ bias,
                      const float* __restrict__ lscale,
                      const float* __restrict__ res,
                      float* __restrict__ out,
                      int Cin) {
  __shared__ float As[16 * ASTR];
  __shared__ float Ys[16 * YSTR];
  __shared__ float s_time[16], s_sc[16];

  int tid  = threadIdx.x;
  int row0 = blockIdx.x * 16;

  // Stage gelu(Xin) tile into LDS (coalesced over k)
  for (int t = tid; t < 16 * Cin; t += 128) {
    int m = t / Cin, k2 = t - m * Cin;
    As[m * ASTR + k2] = gelu_tanh(Xin[(size_t)(row0 + m) * Cin + k2]);
  }
  __syncthreads();

  int w    = tid >> 5;
  int lane = tid & 31;
  int m    = lane & 15;
  int koff = (lane >> 4) << 1;          // lanes 0-15: K+0/K+1; 16-31: K+2/K+3
  int mrow = (lane < 16) ? 0 : 8;       // C/D: VGPR r holds rows r / r+8

  for (int it = 0; it < 3; ++it) {
    int nb   = (w * 3 + it) * 16;
    int ncol = nb + m;
    const float* wrow = Wt + (size_t)ncol * Cin;   // B[k][n] = W[n][k]
    v8f acc = {};
    for (int kb = 0; kb < Cin; kb += 4) {
      v2f a, bb;
      a.x  = As[m * ASTR + kb + koff];
      a.y  = As[m * ASTR + kb + koff + 1];
      bb.x = wrow[kb + koff];
      bb.y = wrow[kb + koff + 1];
      acc = __builtin_amdgcn_wmma_f32_16x16x4_f32(false, a, false, bb,
                                                  (short)0, acc, false, false);
    }
    float bsum = bias[ncol];
#pragma unroll
    for (int r = 0; r < 8; ++r)
      Ys[(r + mrow) * YSTR + nb + m] = acc[r] + bsum;
  }
  __syncthreads();

  // Row-wise Lorentz normalization
  if (tid < 16) {
    float es  = expf(lscale[0]);
    float y0  = Ys[tid * YSTR];
    float tme = es / (1.0f + expf(-y0)) + 1.1f;   // sigmoid(y0)*exp(s)+1.1
    float den = 0.0f;
    for (int c = 1; c < CH; ++c) { float v = Ys[tid * YSTR + c]; den += v * v; }
    den = fmaxf(den, 1e-8f);
    s_time[tid] = tme;
    s_sc[tid]   = sqrtf((tme * tme - 1.0f) / den);
  }
  __syncthreads();

  for (int t = tid; t < 16 * CH; t += 128) {
    int mm = t / CH, c = t - mm * CH;
    size_t o = (size_t)(row0 + mm) * CH + c;
    float v = (c == 0) ? s_time[mm] : Ys[mm * YSTR + c] * s_sc[mm];
    if (res) v += res[o];
    out[o] = v;
  }
}

// ---------------------------------------------------------------------------
// 3) fp32 -> bf16 (RNE) + row squared norm of the *rounded* values so that
//    the WMMA self-distance d[i][i] evaluates to ~0 exactly.
// ---------------------------------------------------------------------------
__global__ void __launch_bounds__(192)
bf16_sq_kernel(const float* __restrict__ x1,
               unsigned short* __restrict__ xb,
               float* __restrict__ sq) {
  __shared__ float red[CH];
  size_t row = blockIdx.x;
  int c = threadIdx.x;
  float v = x1[row * CH + c];
  unsigned int u = __float_as_uint(v);
  unsigned int r = (u + 0x7FFFu + ((u >> 16) & 1u)) >> 16;   // RNE to bf16
  xb[row * CH + c] = (unsigned short)r;
  float fr = __uint_as_float(r << 16);
  red[c] = fr * fr;
  __syncthreads();
  for (int s = 96; s >= 6; s >>= 1) {
    if (c < s) red[c] += red[c + s];
    __syncthreads();
  }
  if (c == 0) {
    float t = 0.0f;
    for (int q = 0; q < 6; ++q) t += red[q];
    sq[row] = t;
  }
}

// ---------------------------------------------------------------------------
// 4) Fused kNN: distance tiles via V_WMMA_F32_16X16X32_BF16, running top-9
//    per row in registers (never materialize the 3136x3136 matrix).
//    Block = 128 threads = 4 waves; each wave owns 16 query rows.
// ---------------------------------------------------------------------------
__global__ void __launch_bounds__(128)
knn_kernel(const unsigned short* __restrict__ xb,
           const float* __restrict__ sq,
           int* __restrict__ idxout) {
  __shared__ float dt[4][16][17];
  int tid  = threadIdx.x;
  int w    = tid >> 5;
  int lane = tid & 31;
  int b    = blockIdx.y;
  int ibase = (blockIdx.x * 4 + w) * 16;

  const unsigned short* xbB = xb + (size_t)b * NPT * CH;
  const float*          sqB = sq + (size_t)b * NPT;

  int  m  = lane & 15;
  bool hi = lane >= 16;
  int  mrow = hi ? 8 : 0;

  // A fragments (16x32 bf16 per K-chunk): per-lane halves live in two
  // contiguous 8-half runs (lanes 0-15: K+0..7 / K+16..23, hi: +8 / +24).
  union AB { uint4 u[2]; v16bf v; };
  AB afr[6];
  {
    const unsigned short* arow = xbB + (size_t)(ibase + m) * CH;
#pragma unroll
    for (int kc = 0; kc < 6; ++kc) {
      int o0 = kc * 32 + (hi ? 8 : 0);
      int o1 = kc * 32 + (hi ? 24 : 16);
      afr[kc].u[0] = *(const uint4*)(arow + o0);
      afr[kc].u[1] = *(const uint4*)(arow + o1);
    }
  }
  float sqi[8];
#pragma unroll
  for (int r = 0; r < 8; ++r) sqi[r] = sqB[ibase + mrow + r];

  float bd[KNN]; int bi[KNN];
#pragma unroll
  for (int t = 0; t < KNN; ++t) { bd[t] = 3.4e38f; bi[t] = 0; }

  for (int jt = 0; jt < NPT / 16; ++jt) {
    int jbase = jt * 16;
    // B fragment (32x16): lane n=m, same K order as A (lo: K+0..7 / K+16..23, hi: +8 / +24)
    const unsigned short* brow = xbB + (size_t)(jbase + m) * CH + (hi ? 8 : 0);
    v8f acc = {};
#pragma unroll
    for (int kc = 0; kc < 6; ++kc) {
      AB bfr;
      bfr.u[0] = *(const uint4*)(brow + kc * 32);
      bfr.u[1] = *(const uint4*)(brow + kc * 32 + 16);
      acc = __builtin_amdgcn_wmma_f32_16x16x32_bf16(false, afr[kc].v, false, bfr.v,
                                                    (short)0, acc, false, false);
    }
    float sqj = sqB[jbase + m];
#pragma unroll
    for (int r = 0; r < 8; ++r)
      dt[w][mrow + r][m] = sqi[r] + sqj - 2.0f * acc[r];
    __syncthreads();

    if (lane < 16) {            // lane L owns query row ibase+L
      for (int c2 = 0; c2 < 16; ++c2) {
        float dd = dt[w][lane][c2];
        if (dd < bd[KNN - 1]) { // sorted insertion; strict < keeps lower index on ties
          int jj = jbase + c2;
#pragma unroll
          for (int p = 0; p < KNN; ++p) {
            if (dd < bd[p]) {
              float td = bd[p]; int tj = bi[p];
              bd[p] = dd; bi[p] = jj;
              dd = td; jj = tj;
            }
          }
        }
      }
    }
    __syncthreads();
  }

  if (lane < 16) {
    int* op = idxout + ((size_t)b * NPT + ibase + lane) * KNN;
#pragma unroll
    for (int t = 0; t < KNN; ++t) op[t] = bi[t];
  }
}

// ---------------------------------------------------------------------------
// 5) Neighbor gather + max message + concat:  h = [x, max_j(x[idx_j]-x)]
// ---------------------------------------------------------------------------
__global__ void __launch_bounds__(192)
gather_kernel(const float* __restrict__ x1,
              const int* __restrict__ idx,
              float* __restrict__ h) {
  size_t row = blockIdx.x;
  int c = threadIdx.x;
  int b = (int)(row / NPT);
  size_t base = (size_t)b * NPT;
  float xc = x1[row * CH + c];
  const int* ip = idx + row * KNN;
  float mx = -3.4e38f;
#pragma unroll
  for (int t = 0; t < KNN; ++t) {
    int j = ip[t];
    float v = x1[(base + (size_t)j) * CH + c];
    mx = fmaxf(mx, v - xc);
  }
  h[row * (2 * CH) + c]      = xc;
  h[row * (2 * CH) + CH + c] = mx;
}

// ---------------------------------------------------------------------------
// 6) Final: out(B,C,HW) = transpose(x3) + x   (shortcut read in native layout)
// ---------------------------------------------------------------------------
__global__ void __launch_bounds__(256)
final_out_kernel(const float* __restrict__ x3,
                 const float* __restrict__ x,
                 float* __restrict__ out) {
  __shared__ float tile[16][17];
  int hw0 = blockIdx.x * 16, c0 = blockIdx.y * 16, b = blockIdx.z;
  int tx = threadIdx.x, ty = threadIdx.y;
  tile[ty][tx] = x3[((size_t)b * HW + hw0 + ty) * CH + c0 + tx];
  __syncthreads();
  size_t o = ((size_t)b * CH + c0 + ty) * HW + hw0 + tx;
  out[o] = tile[tx][ty] + x[o];
}

// ---------------------------------------------------------------------------
extern "C" void kernel_launch(void* const* d_in, const int* in_sizes, int n_in,
                              void* d_out, int out_size, void* d_ws, size_t ws_size,
                              hipStream_t stream) {
  (void)in_sizes; (void)n_in; (void)out_size; (void)ws_size;

  const float* x  = (const float*)d_in[0];
  const float* W1 = (const float*)d_in[1];
  const float* b1 = (const float*)d_in[2];
  const float* s1 = (const float*)d_in[3];
  const float* W2 = (const float*)d_in[4];
  const float* b2 = (const float*)d_in[5];
  const float* s2 = (const float*)d_in[6];
  const float* Wg = (const float*)d_in[7];
  const float* bg = (const float*)d_in[8];
  const float* sg = (const float*)d_in[9];
  const float* W3 = (const float*)d_in[10];
  const float* b3 = (const float*)d_in[11];
  const float* s3 = (const float*)d_in[12];
  const float* W4 = (const float*)d_in[13];
  const float* b4 = (const float*)d_in[14];
  const float* s4 = (const float*)d_in[15];
  // d_in[16] = k (always 9 per setup_inputs)

  const size_t RN = (size_t)BATCH * NPT;          // 25088 token rows
  float* ws = (float*)d_ws;                       // ~126 MB of scratch
  float* xn = ws;                                 // (RN, C)
  float* t1 = xn + RN * CH;                       // (RN, C)
  float* x1 = t1 + RN * CH;                       // (RN, C)   ffn1 output
  float* x2 = x1 + RN * CH;                       // (RN, C)   graph output
  float* hb = x2 + RN * CH;                       // (RN, 2C)  concat
  float* sq = hb + RN * 2 * CH;                   // (RN,)
  unsigned short* xb = (unsigned short*)(sq + RN);  // (RN, C) bf16, 32B aligned
  int* idx = (int*)(xb + RN * CH);                // (RN, 9)

  dim3 tb(16, 16);
  dim3 tg(HW / 16, CH / 16, BATCH);
  hipLaunchKernelGGL(transpose_in_kernel, tg, tb, 0, stream, x, xn);

  const int rowTiles = (int)(RN / 16);            // 1568

  // ffn_lorentz #1
  hipLaunchKernelGGL(lorentz_linear_kernel, dim3(rowTiles), dim3(128), 0, stream,
                     xn, W1, b1, s1, (const float*)nullptr, t1, CH);
  hipLaunchKernelGGL(lorentz_linear_kernel, dim3(rowTiles), dim3(128), 0, stream,
                     t1, W2, b2, s2, xn, x1, CH);

  // graph_conv
  hipLaunchKernelGGL(bf16_sq_kernel, dim3((unsigned)RN), dim3(CH), 0, stream,
                     x1, xb, sq);
  hipLaunchKernelGGL(knn_kernel, dim3(NPT / 64, BATCH), dim3(128), 0, stream,
                     xb, sq, idx);
  hipLaunchKernelGGL(gather_kernel, dim3((unsigned)RN), dim3(CH), 0, stream,
                     x1, idx, hb);
  hipLaunchKernelGGL(lorentz_linear_kernel, dim3(rowTiles), dim3(128), 0, stream,
                     hb, Wg, bg, sg, (const float*)nullptr, x2, 2 * CH);

  // ffn_lorentz #2 (x3 reuses xn; original shortcut re-read from x directly)
  hipLaunchKernelGGL(lorentz_linear_kernel, dim3(rowTiles), dim3(128), 0, stream,
                     x2, W3, b3, s3, (const float*)nullptr, t1, CH);
  hipLaunchKernelGGL(lorentz_linear_kernel, dim3(rowTiles), dim3(128), 0, stream,
                     t1, W4, b4, s4, x2, xn, CH);

  hipLaunchKernelGGL(final_out_kernel, tg, tb, 0, stream, xn, x, (float*)d_out);
}
